// DeepseekQuantizedMLP_44899588112791
// MI455X (gfx1250) — matrix operand order repaired and checked by reading, not compiled, not run on hardware
//
#include <hip/hip_runtime.h>
#include <stdint.h>

// ---------------------------------------------------------------------------
// QuIP#-style quantized SwiGLU MLP for MI455X (gfx1250, wave32, WMMA).
//   Kernel 0: xbf = bf16(x)                                  -> ws   [M, H]
//   Kernel 1: h = silu(x @ Wg^T) * (x @ Wu^T)  (bf16)        -> ws   [M, I]
//   Kernel 2: out = h @ Wd^T                   (f32)         -> out  [M, H]
// Weights dequantized on the fly from the 2MB L2-resident codebook into LDS
// as bf16; MACs via v_wmma_f32_16x16x32_bf16 with f32 accumulators.
// Double-buffered LDS; A-tiles staged with GLOBAL_LOAD_ASYNC_TO_LDS_B128
// (ASYNCcnt).  bf16 conversion expressed as native fptrunc so the backend
// can select the hardware packed convert.
// ---------------------------------------------------------------------------

typedef __attribute__((ext_vector_type(16))) __bf16 v16bf;
typedef __attribute__((ext_vector_type(2)))  __bf16 v2bf;
typedef __attribute__((ext_vector_type(8)))  float  v8f;
typedef __attribute__((ext_vector_type(4)))  int    v4i;

#define HDIM 4096
#define IDIM 11008
#define MTOK 8192
#define CSZ  8

#define BM 128
#define BN 64
#define BK 32
#define NK1 (HDIM / BK)   // 128
#define NK2 (IDIM / BK)   // 344

#if defined(__has_builtin)
#if __has_builtin(__builtin_amdgcn_global_load_async_to_lds_b128) && \
    __has_builtin(__builtin_amdgcn_s_wait_asynccnt)
#define HAVE_ASYNC 1
#endif
#if __has_builtin(__builtin_amdgcn_cvt_pk_bf16_f32)
#define HAVE_CVTPK 1
#endif
#endif

#ifdef HAVE_ASYNC
typedef __attribute__((address_space(1))) void GASV;   // global void
typedef __attribute__((address_space(3))) void LASV;   // LDS void
typedef __attribute__((address_space(1))) v4i  GV4;    // global int4
typedef __attribute__((address_space(3))) v4i  LV4;    // LDS int4
// 16-byte async copy global -> LDS, tracked by ASYNCcnt
#define ASYNC_CP16(dst, src) \
    __builtin_amdgcn_global_load_async_to_lds_b128( \
        (GV4*)(GASV*)(void*)(src), (LV4*)(LASV*)(void*)(dst), 0, 0)
#define ASYNC_WAIT() __builtin_amdgcn_s_wait_asynccnt(0)
#endif

__device__ __forceinline__ unsigned short f2bf(float f) {
    __bf16 h = (__bf16)f;                       // fptrunc, RNE
    return __builtin_bit_cast(unsigned short, h);
}
__device__ __forceinline__ unsigned int pack2bf(float a, float b) {
#ifdef HAVE_CVTPK
    return __builtin_bit_cast(unsigned int, __builtin_amdgcn_cvt_pk_bf16_f32(a, b));
#else
    v2bf t = { (__bf16)a, (__bf16)b };          // native fptrunc pair
    return __builtin_bit_cast(unsigned int, t);
#endif
}

union FragB { uint4 q[2]; v16bf v; };

// ---------------------------------------------------------------------------
// Kernel 0: one-shot f32 -> bf16 conversion of x (memory-bound).
// ---------------------------------------------------------------------------
__global__ __launch_bounds__(256)
void xconvert_kernel(const float* __restrict__ x, unsigned short* __restrict__ xbf) {
    const size_t base = ((size_t)blockIdx.x * 256 + threadIdx.x) * 8;
    float4 a = *(const float4*)(x + base);
    float4 b = *(const float4*)(x + base + 4);
    uint4 o;
    o.x = pack2bf(a.x, a.y);
    o.y = pack2bf(a.z, a.w);
    o.z = pack2bf(b.x, b.y);
    o.w = pack2bf(b.z, b.w);
    *(uint4*)(xbf + base) = o;
}

// ---------------------------------------------------------------------------
// Kernel 1: gate/up projections + SwiGLU -> bf16 h.
// grid = (IDIM/BN=172, MTOK/BM=64), block = 256 (8 waves, 4x2 over 128x64).
// ---------------------------------------------------------------------------
__global__ __launch_bounds__(256)
void qmlp_gateup_kernel(const unsigned short* __restrict__ xbf,   // [MTOK, HDIM] bf16
                        const float* __restrict__ codebook,       // [65536, 8]
                        const int*   __restrict__ idx_gate,       // [IDIM, HDIM/8]
                        const int*   __restrict__ idx_up,         // [IDIM, HDIM/8]
                        unsigned short* __restrict__ hbuf)        // [MTOK, IDIM] bf16
{
    __shared__ __align__(16) unsigned short xs[2][BM * BK];
    __shared__ __align__(16) unsigned short wgs[2][BN * BK];
    __shared__ __align__(16) unsigned short wus[2][BN * BK];

    const int tid   = threadIdx.x;
    const int lane  = tid & 31;
    const int wave  = tid >> 5;
    const int waveM = wave >> 1;      // 0..3
    const int waveN = wave & 1;       // 0..1
    const int halfLane = lane & 15;
    const int hiLane   = lane >> 4;

    const int m0 = blockIdx.y * BM;
    const int n0 = blockIdx.x * BN;

    // loader roles
    const int xrow  = tid >> 1;            // 0..127
    const int xhalf = (tid & 1) * 16;      // 0 / 16
    const int wrow  = tid >> 2;            // 0..63
    const int wcol  = tid & 3;             // 0..3
    const size_t gbase = (size_t)(n0 + wrow) * (HDIM / CSZ);
    const unsigned short* xsrc = xbf + (size_t)(m0 + xrow) * HDIM + xhalf;
    const int xso = xrow * BK + xhalf;     // LDS offset of this thread's x chunk
    const int wso = wrow * BK + wcol * 8;  // LDS offset of this thread's w chunk

    v8f accg[2][2] = {};
    v8f accu[2][2] = {};

    // ---- prologue: stage 0 ---------------------------------------------
    {
#ifdef HAVE_ASYNC
        ASYNC_CP16(&xs[0][xso],     xsrc);
        ASYNC_CP16(&xs[0][xso + 8], xsrc + 8);
#else
        uint4 q0 = ((const uint4*)xsrc)[0];
        uint4 q1 = ((const uint4*)xsrc)[1];
        ((uint4*)&xs[0][xso])[0] = q0;
        ((uint4*)&xs[0][xso])[1] = q1;
#endif
        const unsigned int igv = (unsigned int)idx_gate[gbase + wcol];
        const unsigned int iuv = (unsigned int)idx_up[gbase + wcol];
        const float4* cg = (const float4*)(codebook + (size_t)igv * CSZ);
        const float4* cu = (const float4*)(codebook + (size_t)iuv * CSZ);
        float4 g0 = cg[0], g1 = cg[1], u0 = cu[0], u1 = cu[1];
        unsigned int* dg = (unsigned int*)&wgs[0][wso];
        dg[0] = pack2bf(g0.x, g0.y); dg[1] = pack2bf(g0.z, g0.w);
        dg[2] = pack2bf(g1.x, g1.y); dg[3] = pack2bf(g1.z, g1.w);
        unsigned int* du = (unsigned int*)&wus[0][wso];
        du[0] = pack2bf(u0.x, u0.y); du[1] = pack2bf(u0.z, u0.w);
        du[2] = pack2bf(u1.x, u1.y); du[3] = pack2bf(u1.z, u1.w);
#ifdef HAVE_ASYNC
        ASYNC_WAIT();
#endif
    }
    __syncthreads();

    // ---- main loop (one barrier per K-step, double-buffered) ------------
    for (int kt = 0; kt < NK1; ++kt) {
        const int cur = kt & 1;
        const int nxt = cur ^ 1;
        const bool more = (kt + 1) < NK1;

        float4 g0, g1, u0, u1;
#ifndef HAVE_ASYNC
        uint4 xq0, xq1;
#endif
        if (more) {
            const int k0n = (kt + 1) * BK;
#ifdef HAVE_ASYNC
            ASYNC_CP16(&xs[nxt][xso],     xsrc + k0n);
            ASYNC_CP16(&xs[nxt][xso + 8], xsrc + k0n + 8);
#else
            xq0 = ((const uint4*)(xsrc + k0n))[0];
            xq1 = ((const uint4*)(xsrc + k0n))[1];
#endif
            const int kidx = (k0n >> 3) + wcol;
            const unsigned int igv = (unsigned int)idx_gate[gbase + kidx];
            const unsigned int iuv = (unsigned int)idx_up[gbase + kidx];
            const float4* cg = (const float4*)(codebook + (size_t)igv * CSZ);
            const float4* cu = (const float4*)(codebook + (size_t)iuv * CSZ);
            g0 = cg[0]; g1 = cg[1]; u0 = cu[0]; u1 = cu[1];
        }

        // ---- WMMA from buffer `cur` ---------------------------------
        v16bf afrag[2];
        #pragma unroll
        for (int mi = 0; mi < 2; ++mi) {
            const int r = waveM * 32 + mi * 16 + halfLane;
            FragB ua;
            ua.q[0] = *(const uint4*)&xs[cur][r * BK + hiLane * 8];
            ua.q[1] = *(const uint4*)&xs[cur][r * BK + 16 + hiLane * 8];
            afrag[mi] = ua.v;
        }
        v16bf bg[2], bu[2];
        #pragma unroll
        for (int ni = 0; ni < 2; ++ni) {
            const int c = waveN * 32 + ni * 16 + halfLane;
            FragB ub;
            ub.q[0] = *(const uint4*)&wgs[cur][c * BK + hiLane * 8];
            ub.q[1] = *(const uint4*)&wgs[cur][c * BK + 16 + hiLane * 8];
            bg[ni] = ub.v;
            FragB uu;
            uu.q[0] = *(const uint4*)&wus[cur][c * BK + hiLane * 8];
            uu.q[1] = *(const uint4*)&wus[cur][c * BK + 16 + hiLane * 8];
            bu[ni] = uu.v;
        }
        #pragma unroll
        for (int mi = 0; mi < 2; ++mi)
            #pragma unroll
            for (int ni = 0; ni < 2; ++ni) {
                accg[mi][ni] = __builtin_amdgcn_wmma_f32_16x16x32_bf16(
                    false, afrag[mi], false, bg[ni], (short)0, accg[mi][ni], false, false);
                accu[mi][ni] = __builtin_amdgcn_wmma_f32_16x16x32_bf16(
                    false, afrag[mi], false, bu[ni], (short)0, accu[mi][ni], false, false);
            }

        // ---- stage `nxt` ---------------------------------------------
        if (more) {
            unsigned int* dg = (unsigned int*)&wgs[nxt][wso];
            dg[0] = pack2bf(g0.x, g0.y); dg[1] = pack2bf(g0.z, g0.w);
            dg[2] = pack2bf(g1.x, g1.y); dg[3] = pack2bf(g1.z, g1.w);
            unsigned int* du = (unsigned int*)&wus[nxt][wso];
            du[0] = pack2bf(u0.x, u0.y); du[1] = pack2bf(u0.z, u0.w);
            du[2] = pack2bf(u1.x, u1.y); du[3] = pack2bf(u1.z, u1.w);
#ifndef HAVE_ASYNC
            ((uint4*)&xs[nxt][xso])[0] = xq0;
            ((uint4*)&xs[nxt][xso])[1] = xq1;
#else
            ASYNC_WAIT();
#endif
        }
        __syncthreads();
    }

    // ---- fused SwiGLU epilogue -> bf16 h --------------------------------
    #pragma unroll
    for (int mi = 0; mi < 2; ++mi) {
        #pragma unroll
        for (int ni = 0; ni < 2; ++ni) {
            const int col = n0 + waveN * 32 + ni * 16 + halfLane;
            #pragma unroll
            for (int j = 0; j < 8; ++j) {
                const int row = m0 + waveM * 32 + mi * 16 + hiLane * 8 + j;
                float g = accg[mi][ni][j];
                float u = accu[mi][ni][j];
                float s = g * __builtin_amdgcn_rcpf(1.0f + __expf(-g));  // silu
                hbuf[(size_t)row * IDIM + col] = f2bf(s * u);
            }
        }
    }
}

// ---------------------------------------------------------------------------
// Kernel 2: down projection.  out = h @ Wd^T.
// grid = (HDIM/BN=64, MTOK/BM=64), block = 256.
// ---------------------------------------------------------------------------
__global__ __launch_bounds__(256)
void qmlp_down_kernel(const unsigned short* __restrict__ hbuf, // [MTOK, IDIM] bf16
                      const float* __restrict__ codebook,
                      const int*   __restrict__ idx_down,      // [HDIM, IDIM/8]
                      float*       __restrict__ out)           // [MTOK, HDIM]
{
    __shared__ __align__(16) unsigned short hs[2][BM * BK];
    __shared__ __align__(16) unsigned short wds[2][BN * BK];

    const int tid   = threadIdx.x;
    const int lane  = tid & 31;
    const int wave  = tid >> 5;
    const int waveM = wave >> 1;
    const int waveN = wave & 1;
    const int halfLane = lane & 15;
    const int hiLane   = lane >> 4;

    const int m0 = blockIdx.y * BM;
    const int n0 = blockIdx.x * BN;

    const int xrow  = tid >> 1;
    const int xhalf = (tid & 1) * 16;
    const int wrow  = tid >> 2;
    const int wcol  = tid & 3;
    const size_t gbase = (size_t)(n0 + wrow) * (IDIM / CSZ);
    const unsigned short* hsrc = hbuf + (size_t)(m0 + xrow) * IDIM + xhalf;
    const int xso = xrow * BK + xhalf;
    const int wso = wrow * BK + wcol * 8;

    v8f acc[2][2] = {};

    // ---- prologue: stage 0 ---------------------------------------------
    {
#ifdef HAVE_ASYNC
        ASYNC_CP16(&hs[0][xso],     hsrc);
        ASYNC_CP16(&hs[0][xso + 8], hsrc + 8);
#else
        uint4 q0 = ((const uint4*)hsrc)[0];
        uint4 q1 = ((const uint4*)hsrc)[1];
        ((uint4*)&hs[0][xso])[0] = q0;
        ((uint4*)&hs[0][xso])[1] = q1;
#endif
        const unsigned int idv = (unsigned int)idx_down[gbase + wcol];
        const float4* cd = (const float4*)(codebook + (size_t)idv * CSZ);
        float4 d0 = cd[0], d1 = cd[1];
        unsigned int* dd = (unsigned int*)&wds[0][wso];
        dd[0] = pack2bf(d0.x, d0.y); dd[1] = pack2bf(d0.z, d0.w);
        dd[2] = pack2bf(d1.x, d1.y); dd[3] = pack2bf(d1.z, d1.w);
#ifdef HAVE_ASYNC
        ASYNC_WAIT();
#endif
    }
    __syncthreads();

    for (int kt = 0; kt < NK2; ++kt) {
        const int cur = kt & 1;
        const int nxt = cur ^ 1;
        const bool more = (kt + 1) < NK2;

        float4 d0, d1;
#ifndef HAVE_ASYNC
        uint4 hq0, hq1;
#endif
        if (more) {
            const int k0n = (kt + 1) * BK;
#ifdef HAVE_ASYNC
            ASYNC_CP16(&hs[nxt][xso],     hsrc + k0n);
            ASYNC_CP16(&hs[nxt][xso + 8], hsrc + k0n + 8);
#else
            hq0 = ((const uint4*)(hsrc + k0n))[0];
            hq1 = ((const uint4*)(hsrc + k0n))[1];
#endif
            const int kidx = (k0n >> 3) + wcol;
            const unsigned int idv = (unsigned int)idx_down[gbase + kidx];
            const float4* cd = (const float4*)(codebook + (size_t)idv * CSZ);
            d0 = cd[0]; d1 = cd[1];
        }

        v16bf afrag[2];
        #pragma unroll
        for (int mi = 0; mi < 2; ++mi) {
            const int r = waveM * 32 + mi * 16 + halfLane;
            FragB ua;
            ua.q[0] = *(const uint4*)&hs[cur][r * BK + hiLane * 8];
            ua.q[1] = *(const uint4*)&hs[cur][r * BK + 16 + hiLane * 8];
            afrag[mi] = ua.v;
        }
        v16bf bfrag[2];
        #pragma unroll
        for (int ni = 0; ni < 2; ++ni) {
            const int c = waveN * 32 + ni * 16 + halfLane;
            FragB ub;
            ub.q[0] = *(const uint4*)&wds[cur][c * BK + hiLane * 8];
            ub.q[1] = *(const uint4*)&wds[cur][c * BK + 16 + hiLane * 8];
            bfrag[ni] = ub.v;
        }
        #pragma unroll
        for (int mi = 0; mi < 2; ++mi)
            #pragma unroll
            for (int ni = 0; ni < 2; ++ni)
                acc[mi][ni] = __builtin_amdgcn_wmma_f32_16x16x32_bf16(
                    false, afrag[mi], false, bfrag[ni], (short)0, acc[mi][ni], false, false);

        if (more) {
            unsigned int* dd = (unsigned int*)&wds[nxt][wso];
            dd[0] = pack2bf(d0.x, d0.y); dd[1] = pack2bf(d0.z, d0.w);
            dd[2] = pack2bf(d1.x, d1.y); dd[3] = pack2bf(d1.z, d1.w);
#ifndef HAVE_ASYNC
            ((uint4*)&hs[nxt][xso])[0] = hq0;
            ((uint4*)&hs[nxt][xso])[1] = hq1;
#else
            ASYNC_WAIT();
#endif
        }
        __syncthreads();
    }

    #pragma unroll
    for (int mi = 0; mi < 2; ++mi) {
        #pragma unroll
        for (int ni = 0; ni < 2; ++ni) {
            const int col = n0 + waveN * 32 + ni * 16 + halfLane;
            #pragma unroll
            for (int j = 0; j < 8; ++j) {
                const int row = m0 + waveM * 32 + mi * 16 + hiLane * 8 + j;
                out[(size_t)row * HDIM + col] = acc[mi][ni][j];
            }
        }
    }
}

// ---------------------------------------------------------------------------
extern "C" void kernel_launch(void* const* d_in, const int* in_sizes, int n_in,
                              void* d_out, int out_size, void* d_ws, size_t ws_size,
                              hipStream_t stream) {
    const float* x        = (const float*)d_in[0];   // [4,2048,4096] f32
    const float* codebook = (const float*)d_in[1];   // [65536,8] f32
    const int*   idx_gate = (const int*)d_in[2];     // [11008,512]
    const int*   idx_up   = (const int*)d_in[3];     // [11008,512]
    const int*   idx_down = (const int*)d_in[4];     // [4096,1376]
    float*       out      = (float*)d_out;           // [4,2048,4096] f32

    unsigned short* xbf  = (unsigned short*)d_ws;                       // 67 MB
    unsigned short* hbuf = (unsigned short*)((char*)d_ws +
                             (size_t)MTOK * HDIM * sizeof(unsigned short)); // 180 MB

    const int cvt_blocks = (MTOK * HDIM) / (8 * 256);   // 16384
    xconvert_kernel<<<cvt_blocks, 256, 0, stream>>>(x, xbf);

    dim3 grid1(IDIM / BN, MTOK / BM);   // (172, 64)
    dim3 grid2(HDIM / BN, MTOK / BM);   // (64, 64)
    qmlp_gateup_kernel<<<grid1, 256, 0, stream>>>(xbf, codebook, idx_gate, idx_up, hbuf);
    qmlp_down_kernel<<<grid2, 256, 0, stream>>>(hbuf, codebook, idx_down, out);
}
